// MVAS_41695542510270
// MI455X (gfx1250) — compile-verified, matrix operand order checked by reading, not run on hardware
//
#include <hip/hip_runtime.h>
#include <hip/hip_bf16.h>
#include <math.h>
#include <stdint.h>

// ---------------------------------------------------------------------------
// Multi-view windowed routing attention for MI455X (gfx1250, wave32, WMMA).
//
// cv (1,128,128,256) f32; mv (1,4,128,128,256) f32
// 8x8 windows of 16x16 px -> p2=64, w2=256 ; heads m=8, ch=32 ; TOPK=4, v=4
// scale = 256^-0.5 = 1/16
//
// f16 WMMA (v_wmma_f32_16x16x32_f16) flash attention, f32 accumulate.
// ch=32 == WMMA K exactly. ~17.2 GFLOP vs ~100MB traffic -> balanced ~4us.
// KV chunks staged LDS via the Tensor Data Mover (double buffered,
// s_wait_tensorcnt), softmax reductions via DPP16 (no LDS round trips).
// ---------------------------------------------------------------------------

typedef __attribute__((ext_vector_type(16))) _Float16 v16h;
typedef __attribute__((ext_vector_type(8)))  float    v8f;
typedef __attribute__((ext_vector_type(4)))  unsigned v4u;
typedef __attribute__((ext_vector_type(8)))  int      v8i;
typedef __attribute__((ext_vector_type(4)))  int      v4i;

union FragH { v16h v; _Float16 h[16]; };
union PackU { unsigned u; _Float16 h[2]; };

#if __has_builtin(__builtin_amdgcn_tensor_load_to_lds)
#define USE_TDM 1
#else
#define USE_TDM 0
#endif

// ---- DPP16 butterfly reduction within each 16-lane half -------------------
// xor1 (quad_perm [1,0,3,2]) ; xor2 (quad_perm [2,3,0,1]) ;
// ROW_HALF_MIRROR completes the 8-group once quads are uniform ;
// ROW_MIRROR completes the 16-group. Pure VALU, no LDS.
template <int CTRL>
__device__ __forceinline__ float dpp_mov(float v) {
  return __int_as_float(
      __builtin_amdgcn_update_dpp(0, __float_as_int(v), CTRL, 0xF, 0xF, true));
}
__device__ __forceinline__ float redmax16(float v) {
  v = fmaxf(v, dpp_mov<0xB1>(v));     // quad_perm [1,0,3,2]
  v = fmaxf(v, dpp_mov<0x4E>(v));     // quad_perm [2,3,0,1]
  v = fmaxf(v, dpp_mov<0x141>(v));    // row_half_mirror
  v = fmaxf(v, dpp_mov<0x140>(v));    // row_mirror
  return v;
}
__device__ __forceinline__ float redsum16(float v) {
  v += dpp_mov<0xB1>(v);
  v += dpp_mov<0x4E>(v);
  v += dpp_mov<0x141>(v);
  v += dpp_mov<0x140>(v);
  return v;
}

// ---------------------------------------------------------------------------
// Kernel 1: window means. blocks 0..63 -> q_win[p][c]; 64..319 -> k_win[q][c]
// ---------------------------------------------------------------------------
__global__ __launch_bounds__(256) void mvas_win_mean(
    const float* __restrict__ cv, const float* __restrict__ mv,
    float* __restrict__ qwin, float* __restrict__ kwin) {
  const int b = blockIdx.x;
  const int c = threadIdx.x;
  if (b < 64) {
    const int jw = b >> 3, iw = b & 7;
    const float* base = cv + (size_t)((jw * 16) * 128 + iw * 16) * 256 + c;
    float s = 0.f;
    for (int hh = 0; hh < 16; ++hh)
      for (int ww = 0; ww < 16; ++ww)
        s += base[(size_t)(hh * 128 + ww) * 256];
    qwin[b * 256 + c] = s * (1.0f / 256.0f);
  } else {
    const int q = b - 64;
    const int vv = q >> 6, pp = q & 63, jw = pp >> 3, iw = pp & 7;
    const float* base = mv + (size_t)((vv * 128 + jw * 16) * 128 + iw * 16) * 256 + c;
    float s = 0.f;
    for (int hh = 0; hh < 16; ++hh)
      for (int ww = 0; ww < 16; ++ww)
        s += base[(size_t)(hh * 128 + ww) * 256];
    kwin[q * 256 + c] = s * (1.0f / 256.0f);
  }
}

// ---------------------------------------------------------------------------
// Kernel 2: routing logits + top-4 (strict '>' == jax.lax.top_k tie rule)
// ---------------------------------------------------------------------------
__global__ __launch_bounds__(256) void mvas_route_topk(
    const float* __restrict__ qwin, const float* __restrict__ kwin,
    int* __restrict__ ridx) {
  __shared__ float qrow[256];
  __shared__ float logits[256];
  const int p = blockIdx.x, t = threadIdx.x;
  qrow[t] = qwin[p * 256 + t] * 0.0625f;
  __syncthreads();
  const float* kr = kwin + t * 256;
  float acc = 0.f;
  for (int c = 0; c < 256; ++c) acc += qrow[c] * kr[c];
  logits[t] = acc;
  __syncthreads();
  if (t == 0) {
    for (int k = 0; k < 4; ++k) {
      int best = 0; float bv = logits[0];
      for (int q = 1; q < 256; ++q) {
        float vq = logits[q];
        if (vq > bv) { bv = vq; best = q; }
      }
      ridx[p * 4 + k] = best;
      logits[best] = -3.4e38f;
    }
  }
}

// ---------------------------------------------------------------------------
// Kernel 3: flash attention per (window, head). 8 waves x 32 Q rows.
// ---------------------------------------------------------------------------
__global__ __launch_bounds__(256) void mvas_attn(
    const float* __restrict__ cv, const float* __restrict__ mv,
    const int* __restrict__ ridx, float* __restrict__ out) {
  __shared__ float    kv32[2][32 * 33];   // f32 KV chunk, 33-dword row stride
  __shared__ unsigned pl32[8][32 * 18];   // packed-f16 P scratch per wave
  __shared__ int      srid[4];

  const int p    = blockIdx.x;            // window 0..63
  const int head = blockIdx.y;            // 0..7
  const int tid  = threadIdx.x;
  const int lane = tid & 31;
  const int wave = tid >> 5;
  const int lane16 = lane & 15;
  const int hi = lane >> 4;
  const int kb = hi << 3;                 // frag K-half base (ISA A/B layout)

  if (tid < 4) srid[tid] = ridx[p * 4 + tid];
  __syncthreads();

  const int jw = p >> 3, iw = p & 7;

  // ---- Q fragments: A operand. lanes<16: h[0..7]=K0..7,h[8..15]=K16..23;
  // lanes>=16: K8..15 / K24..31. scale 1/16 folded in before f16 cvt.
  FragH a0, a1;
#pragma unroll
  for (int g = 0; g < 2; ++g) {
    const int qrow = wave * 32 + g * 16 + lane16;
    const int hh = qrow >> 4, ww = qrow & 15;
    const float* qp = cv + (size_t)((jw * 16 + hh) * 128 + iw * 16 + ww) * 256
                         + head * 32;
    FragH& a = g ? a1 : a0;
#pragma unroll
    for (int i = 0; i < 8; ++i) {
      a.h[i]     = (_Float16)(qp[kb + i]      * 0.0625f);
      a.h[8 + i] = (_Float16)(qp[16 + kb + i] * 0.0625f);
    }
  }

  v8f acc[2][2];
  float mrow[2][8], lrow[2][8];
#pragma unroll
  for (int g = 0; g < 2; ++g)
#pragma unroll
    for (int r = 0; r < 8; ++r) {
      mrow[g][r] = -3.0e38f; lrow[g][r] = 0.f;
      acc[g][0][r] = 0.f; acc[g][1][r] = 0.f;
    }

#if USE_TDM
  // Issue one TDM descriptor per 32kv x 32ch chunk:
  // 3D tile: x = 32 ch (f32), y = 16 px @ stride 256, z = 2 hh @ stride 128*256
  // pad_enable adds 1 dword after every 32 dwords -> LDS row stride 33.
  auto tdm_issue = [&](int t, int buf) {
    const int widx = srid[t >> 3];
    const int vv = widx >> 6, pp = widx & 63;
    const int hh0 = (t & 7) * 2;
    const float* gp = mv
        + (size_t)((vv * 128 + (pp >> 3) * 16 + hh0) * 128 + (pp & 7) * 16) * 256
        + head * 32;
    const unsigned long long ga = (unsigned long long)(uintptr_t)gp;
    const unsigned lds = (unsigned)(uintptr_t)&kv32[buf][0];
    v4u g0;
    g0[0] = 1u;                                   // count=1, gather off
    g0[1] = lds;                                  // lds_addr
    g0[2] = (unsigned)ga;                         // global_addr[31:0]
    g0[3] = (unsigned)((ga >> 32) & 0x1FFFFFFull) // global_addr[56:32]
            | 0x80000000u;                        // type=2 ("image")
    v8i g1;
    g1[0] = (2 << 16)        // data_size = 4B
          | (1 << 20)        // pad_enable
          | (4 << 22);       // pad_interval: 2^(4+1)=32 dwords, pad_amount=1dw
    g1[1] = (int)(32u << 16);            // tensor_dim0 = 32 (low16 in [31:16])
    g1[2] = (int)(16u << 16);            // tensor_dim1 = 16
    g1[3] = (int)(32u << 16);            // tile_dim0 = 32
    g1[4] = (int)(16u | (2u << 16));     // tile_dim1 = 16, tile_dim2 = 2
    g1[5] = 256;                         // tensor_dim0_stride = 256
    g1[6] = (int)0x80000000u;            // tensor_dim1_stride = 32768 (low16)
    g1[7] = 0;
    v4i g2; g2[0] = 2; g2[1] = 0; g2[2] = 0; g2[3] = 0;   // tensor_dim2 = 2
    v4i g3 = {0, 0, 0, 0};
#if __clang_major__ >= 23
    v8i g4 = {0, 0, 0, 0, 0, 0, 0, 0};
    __builtin_amdgcn_tensor_load_to_lds(g0, g1, g2, g3, g4, 0);
#else
    __builtin_amdgcn_tensor_load_to_lds(g0, g1, g2, g3, 0);
#endif
  };
  if (wave == 0) tdm_issue(0, 0);
#endif

  for (int t = 0; t < 32; ++t) {
    const int buf = t & 1;
#if USE_TDM
    if (wave == 0) {
      if (t + 1 < 32) {
        tdm_issue(t + 1, buf ^ 1);               // prefetch next chunk
        __builtin_amdgcn_s_wait_tensorcnt(1);    // chunk t has landed
      } else {
        __builtin_amdgcn_s_wait_tensorcnt(0);
      }
    }
#else
    {   // cooperative fallback staging (f32, row stride 33 dwords)
      const int r  = tid >> 3;
      const int c4 = (tid & 7) << 2;
      const int widx = srid[t >> 3];
      const int pix  = ((t & 7) << 5) + r;
      const int vv = widx >> 6, pp = widx & 63;
      const float* kp = mv
          + (size_t)((vv * 128 + (pp >> 3) * 16 + (pix >> 4)) * 128
                     + (pp & 7) * 16 + (pix & 15)) * 256
          + head * 32 + c4;
      const float4 f = *reinterpret_cast<const float4*>(kp);
      float* dst = &kv32[buf][r * 33 + c4];
      dst[0] = f.x; dst[1] = f.y; dst[2] = f.z; dst[3] = f.w;
      if (t + 1 < 32) {
        const int nw = srid[(t + 1) >> 3];
        const int np = ((t + 1) & 7) * 32 + r;
        const int nvv = nw >> 6, npp = nw & 63;
        __builtin_prefetch(mv + (size_t)((nvv * 128 + (npp >> 3) * 16 + (np >> 4)) * 128
                                         + (npp & 7) * 16 + (np & 15)) * 256
                              + head * 32 + c4, 0, 1);
      }
    }
#endif
    __syncthreads();
    const float* kc = &kv32[buf][0];

    // ---- K as B operand (32x16): lane = kv column, reads K-row ch values
    FragH bk0, bk1;
#pragma unroll
    for (int i = 0; i < 8; ++i) {
      bk0.h[i]     = (_Float16)kc[lane16 * 33 + kb + i];
      bk0.h[8 + i] = (_Float16)kc[lane16 * 33 + 16 + kb + i];
      bk1.h[i]     = (_Float16)kc[(16 + lane16) * 33 + kb + i];
      bk1.h[8 + i] = (_Float16)kc[(16 + lane16) * 33 + 16 + kb + i];
    }

    const v8f z = {};
    v8f s00 = __builtin_amdgcn_wmma_f32_16x16x32_f16(false, a0.v, false, bk0.v, (short)0, z, false, false);
    v8f s01 = __builtin_amdgcn_wmma_f32_16x16x32_f16(false, a0.v, false, bk1.v, (short)0, z, false, false);
    v8f s10 = __builtin_amdgcn_wmma_f32_16x16x32_f16(false, a1.v, false, bk0.v, (short)0, z, false, false);
    v8f s11 = __builtin_amdgcn_wmma_f32_16x16x32_f16(false, a1.v, false, bk1.v, (short)0, z, false, false);

    // ---- online softmax; row lives in one VGPR across a 16-lane half ----
#pragma unroll
    for (int g = 0; g < 2; ++g) {
      v8f& sA = g ? s10 : s00;
      v8f& sB = g ? s11 : s01;
      float corr[8];
#pragma unroll
      for (int r = 0; r < 8; ++r) {
        const float rm = redmax16(fmaxf(sA[r], sB[r]));
        const float mold = mrow[g][r];
        const float mnew = fmaxf(mold, rm);
        const float cr = __expf(mold - mnew);
        const float pa = __expf(sA[r] - mnew);
        const float pb = __expf(sB[r] - mnew);
        sA[r] = pa; sB[r] = pb;
        lrow[g][r] = lrow[g][r] * cr + redsum16(pa + pb);
        mrow[g][r] = mnew;
        corr[r] = cr;
      }
#pragma unroll
      for (int c = 0; c < 2; ++c)
#pragma unroll
        for (int r = 0; r < 8; ++r) acc[g][c][r] *= corr[r];

      // P tile -> wave-private LDS, (kv=c, kv=c+16) packed in one dword
      unsigned* pw = &pl32[wave][0];
#pragma unroll
      for (int r = 0; r < 8; ++r) {
        const int row = g * 16 + r + kb;           // lanes>=16 hold M=r+8
        PackU pk; pk.h[0] = (_Float16)sA[r]; pk.h[1] = (_Float16)sB[r];
        pw[row * 18 + lane16] = pk.u;
      }
    }
    // wave-private scratch: only our own DS writes must land
    asm volatile("s_wait_dscnt 0" ::: "memory");

    // ---- P as A operand, V as B operand ----
    FragH ap0, ap1, bv0, bv1;
    const unsigned* pw = &pl32[wave][0];
#pragma unroll
    for (int i = 0; i < 8; ++i) {
      PackU w0; w0.u = pw[lane16 * 18 + kb + i];
      ap0.h[i] = w0.h[0]; ap0.h[8 + i] = w0.h[1];
      PackU w1; w1.u = pw[(16 + lane16) * 18 + kb + i];
      ap1.h[i] = w1.h[0]; ap1.h[8 + i] = w1.h[1];
      bv0.h[i]     = (_Float16)kc[(kb + i) * 33 + lane16];
      bv0.h[8 + i] = (_Float16)kc[(16 + kb + i) * 33 + lane16];
      bv1.h[i]     = (_Float16)kc[(kb + i) * 33 + 16 + lane16];
      bv1.h[8 + i] = (_Float16)kc[(16 + kb + i) * 33 + 16 + lane16];
    }
    acc[0][0] = __builtin_amdgcn_wmma_f32_16x16x32_f16(false, ap0.v, false, bv0.v, (short)0, acc[0][0], false, false);
    acc[0][1] = __builtin_amdgcn_wmma_f32_16x16x32_f16(false, ap0.v, false, bv1.v, (short)0, acc[0][1], false, false);
    acc[1][0] = __builtin_amdgcn_wmma_f32_16x16x32_f16(false, ap1.v, false, bv0.v, (short)0, acc[1][0], false, false);
    acc[1][1] = __builtin_amdgcn_wmma_f32_16x16x32_f16(false, ap1.v, false, bv1.v, (short)0, acc[1][1], false, false);

    __syncthreads();   // kv buffer reused two chunks later
  }

  // ---- epilogue: normalize and scatter to un-partitioned NHWC output ----
#pragma unroll
  for (int g = 0; g < 2; ++g)
#pragma unroll
    for (int c = 0; c < 2; ++c)
#pragma unroll
      for (int r = 0; r < 8; ++r) {
        const int row = wave * 32 + g * 16 + r + kb;
        const int hh = row >> 4, ww = row & 15;
        const int ch = head * 32 + c * 16 + lane16;
        out[(size_t)((jw * 16 + hh) * 128 + iw * 16 + ww) * 256 + ch] =
            acc[g][c][r] / lrow[g][r];
      }
}

// ---------------------------------------------------------------------------
extern "C" void kernel_launch(void* const* d_in, const int* in_sizes, int n_in,
                              void* d_out, int out_size, void* d_ws, size_t ws_size,
                              hipStream_t stream) {
  (void)in_sizes; (void)n_in; (void)out_size; (void)ws_size;
  const float* cv = (const float*)d_in[0];   // (1,128,128,256)
  const float* mv = (const float*)d_in[1];   // (1,4,128,128,256)
  float* out  = (float*)d_out;               // (1,128,128,256)

  float* qwin = (float*)d_ws;                // 64*256 f32
  float* kwin = qwin + 64 * 256;             // 256*256 f32
  int*   ridx = (int*)(kwin + 256 * 256);    // 64*4 int

  hipLaunchKernelGGL(mvas_win_mean, dim3(320), dim3(256), 0, stream,
                     cv, mv, qwin, kwin);
  hipLaunchKernelGGL(mvas_route_topk, dim3(64), dim3(256), 0, stream,
                     qwin, kwin, ridx);
  hipLaunchKernelGGL(mvas_attn, dim3(64, 8), dim3(256), 0, stream,
                     cv, mv, ridx, out);
}
